// SegFormerXAttention_309237645573
// MI455X (gfx1250) — matrix operand order repaired and checked by reading, not compiled, not run on hardware
//
#include <hip/hip_runtime.h>
#include <hip/hip_bf16.h>

typedef __attribute__((ext_vector_type(16))) __bf16 v16bf;
typedef __attribute__((ext_vector_type(8)))  __bf16 v8bf;
typedef __attribute__((ext_vector_type(8)))  float  v8f;

#define NEGV (-10000.0f)

#if defined(__has_builtin)
#  if __has_builtin(__builtin_amdgcn_tensor_load_to_lds)
#    define USE_TDM 1
#  endif
#endif
#ifndef USE_TDM
#  define USE_TDM 0
#endif

#if USE_TDM
typedef unsigned int u32x4 __attribute__((ext_vector_type(4)));
typedef int          i32x4 __attribute__((ext_vector_type(4)));
typedef int          i32x8 __attribute__((ext_vector_type(8)));
#endif

__device__ __forceinline__ v8f wmma_bf16(v16bf a, v16bf b, v8f c) {
    return __builtin_amdgcn_wmma_f32_16x16x32_bf16(
        false, a, false, b, (short)0, c, false, false);
}

// A-matrix 16x32 bf16 frag: lane half h holds K = h*8..h*8+7 and 16+h*8..+7
__device__ __forceinline__ v16bf load_afrag(const __bf16* rowptr, int k0, int half) {
    union { v16bf v; v8bf h[2]; } u;
    u.h[0] = *(const v8bf*)(rowptr + k0 + half * 8);
    u.h[1] = *(const v8bf*)(rowptr + k0 + 16 + half * 8);
    return u.v;
}

// B-matrix 32x16 bf16 frag from N x K row-major source (global or LDS):
// lane half h holds K = h*8..h*8+7 and 16+h*8..+7
__device__ __forceinline__ v16bf load_bfrag(const __bf16* rowptr, int k0, int half) {
    union { v16bf v; v8bf h[2]; } u;
    u.h[0] = *(const v8bf*)(rowptr + k0 + half * 8);
    u.h[1] = *(const v8bf*)(rowptr + k0 + 16 + half * 8);
    return u.v;
}

__global__ void __launch_bounds__(256)
cvt_kernel(const float* __restrict__ src, __bf16* __restrict__ dst, int n) {
    int i = blockIdx.x * blockDim.x + threadIdx.x;
    if (i < n) dst[i] = (__bf16)src[i];
}

// Stage a 64(row) x 32(col) bf16 tile of W (row stride 1024) into LDS buffer.
// TDM path: wave 0 issues one tensor_load_to_lds descriptor per the CDNA5 D#.
__device__ __forceinline__ void stage_wtile(const __bf16* gsrc, __bf16* lbuf,
                                            int tid, int wave) {
#if USE_TDM
    if (wave == 0) {
        unsigned long long ga = (unsigned long long)(uintptr_t)gsrc;
        unsigned int la = (unsigned int)(uintptr_t)lbuf;  // low 32b = LDS offset
        u32x4 g0;
        g0[0] = 1u;                                   // count=1, user descriptor
        g0[1] = la;                                   // lds_addr
        g0[2] = (unsigned int)ga;                     // global_addr[31:0]
        g0[3] = (unsigned int)((ga >> 32) & 0x01FFFFFFu) | (2u << 30); // [56:32]|type=2
        i32x8 g1;
        g1[0] = (int)(1u << 16);                      // data_size=1 (2 bytes)
        g1[1] = (int)(1024u << 16);                   // tensor_dim0 = 1024 (lo16)
        g1[2] = (int)(64u << 16);                     // dim0 hi=0 | tensor_dim1=64
        g1[3] = (int)(32u << 16);                     // dim1 hi=0 | tile_dim0=32
        g1[4] = (int)(64u);                           // tile_dim1=64, tile_dim2=0
        g1[5] = (int)1024;                            // tensor_dim0_stride = 1024
        g1[6] = 0;                                    // stride hi / dim1_stride
        g1[7] = 0;
        i32x4 z4 = {0, 0, 0, 0};
        i32x8 z8 = {0, 0, 0, 0, 0, 0, 0, 0};
        __builtin_amdgcn_tensor_load_to_lds(g0, g1, z4, z4, z8, 0);
    }
#else
    int row = tid >> 2, col = (tid & 3) * 8;          // 256 thr * 8 elems = 64x32
    *(v8bf*)(lbuf + row * 32 + col) = *(const v8bf*)(gsrc + (size_t)row * 1024 + col);
#endif
}

__device__ __forceinline__ void wait_stage(int wave) {
#if USE_TDM
    if (wave == 0) __builtin_amdgcn_s_wait_tensorcnt(0);
#endif
    __syncthreads();
}

// C[M,N] = X[M,1024] * W[N,1024]^T + bias; out in bf16.
// Block: 8 waves, 256(M) x 64(N). Wave: 32(M) x 64(N). W tile staged in LDS.
// layout 0: out[(b*16+h)*L*64 + l*64 + dh]          (B,H,L,64)
// layout 1: out[((b*16+h)*64 + dh)*1280 + kpos+l]   (B,H,64,1280)  (V transposed)
__global__ void __launch_bounds__(256)
proj_gemm(const __bf16* __restrict__ X, const __bf16* __restrict__ W,
          const float* __restrict__ bias, __bf16* __restrict__ out,
          int L, int layout, int kpos_off)
{
    __shared__ __bf16 wtile[2][64 * 32];
    const int tid = threadIdx.x;
    const int lane = tid & 31;
    const int wave = tid >> 5;
    const int half = lane >> 4, l16 = lane & 15;
    const int m0 = blockIdx.x * 256 + wave * 32;
    const int n0 = blockIdx.y * 64;

    const __bf16* arow0 = X + (size_t)(m0 + l16) * 1024;
    const __bf16* arow1 = X + (size_t)(m0 + 16 + l16) * 1024;
    const __bf16* wslab = W + (size_t)n0 * 1024;

    v8f acc[8];
#pragma unroll
    for (int t = 0; t < 8; ++t) acc[t] = (v8f)0.0f;

    stage_wtile(wslab, wtile[0], tid, wave);
    wait_stage(wave);

    for (int k = 0; k < 32; ++k) {
        const int cur = k & 1;
        const int k0 = k * 32;
        if (k < 31)
            stage_wtile(wslab + (k0 + 32), wtile[cur ^ 1], tid, wave);

        v16bf a0 = load_afrag(arow0, k0, half);
        v16bf a1 = load_afrag(arow1, k0, half);
#pragma unroll
        for (int t = 0; t < 4; ++t) {
            v16bf b = load_bfrag(wtile[cur] + (t * 16 + l16) * 32, 0, half);
            acc[t]     = wmma_bf16(a0, b, acc[t]);
            acc[4 + t] = wmma_bf16(a1, b, acc[4 + t]);
        }
        wait_stage(wave);
    }

#pragma unroll
    for (int g = 0; g < 2; ++g) {
#pragma unroll
        for (int t = 0; t < 4; ++t) {
            int n = n0 + t * 16 + l16;
            float bs = bias[n];
            int h = n >> 6, dh = n & 63;
#pragma unroll
            for (int r = 0; r < 8; ++r) {
                int m = m0 + g * 16 + r + 8 * half;
                int bb = m / L, l = m - bb * L;
                float v = acc[g * 4 + t][r] + bs;
                size_t idx;
                if (layout == 0)
                    idx = ((size_t)(bb * 16 + h) * L + l) * 64 + dh;
                else
                    idx = ((size_t)(bb * 16 + h) * 64 + dh) * 1280 + (kpos_off + l);
                out[idx] = (__bf16)v;
            }
        }
    }
}

// One wave handles 16 query rows for one (b,h). Flash-style online softmax
// over 1024 vid keys (Q1 vs Kv) then 256 txt keys (Q2 vs Kt).
__global__ void __launch_bounds__(32)
attn_kernel(const __bf16* __restrict__ Q1, const __bf16* __restrict__ Q2,
            const __bf16* __restrict__ Kv, const __bf16* __restrict__ Kt,
            const __bf16* __restrict__ Vt,
            const unsigned char* __restrict__ mq,
            const unsigned char* __restrict__ mkv,
            const unsigned char* __restrict__ mkt,
            float* __restrict__ out, int Lq)
{
    __shared__ __bf16 lds[16 * 32];
    const int lane = threadIdx.x & 31;
    const int half = lane >> 4, l16 = lane & 15;
    const int q0 = blockIdx.x * 16;
    const int h = blockIdx.y, b = blockIdx.z;

    const __bf16* q1row = Q1 + ((size_t)(b * 16 + h) * Lq + q0 + l16) * 64;
    const __bf16* q2row = Q2 + ((size_t)(b * 16 + h) * Lq + q0 + l16) * 64;
    v16bf qf1[2], qf2[2];
    qf1[0] = load_afrag(q1row, 0, half);  qf1[1] = load_afrag(q1row, 32, half);
    qf2[0] = load_afrag(q2row, 0, half);  qf2[1] = load_afrag(q2row, 32, half);

    bool mrow[8];
#pragma unroll
    for (int r = 0; r < 8; ++r)
        mrow[r] = mq[b * Lq + q0 + r + 8 * half] != 0;

    float mi[8], li[8];
#pragma unroll
    for (int r = 0; r < 8; ++r) { mi[r] = -__builtin_inff(); li[r] = 0.0f; }
    v8f o[4];
#pragma unroll
    for (int t = 0; t < 4; ++t) o[t] = (v8f)0.0f;

    const __bf16* vbase = Vt + (size_t)(b * 16 + h) * 64 * 1280;

    auto tile = [&](const v16bf* qf, const __bf16* Kbase,
                    const unsigned char* mk, int kb, int kpos) {
        const __bf16* krow0 = Kbase + (size_t)(kb + l16) * 64;
        const __bf16* krow1 = Kbase + (size_t)(kb + 16 + l16) * 64;
        v8f s0 = (v8f)0.0f, s1 = (v8f)0.0f;
        s0 = wmma_bf16(qf[0], load_bfrag(krow0, 0, half), s0);
        s0 = wmma_bf16(qf[1], load_bfrag(krow0, 32, half), s0);
        s1 = wmma_bf16(qf[0], load_bfrag(krow1, 0, half), s1);
        s1 = wmma_bf16(qf[1], load_bfrag(krow1, 32, half), s1);
        bool mk0 = mk[kb + l16] != 0;
        bool mk1 = mk[kb + 16 + l16] != 0;
#pragma unroll
        for (int r = 0; r < 8; ++r) {
            float a = (mrow[r] && mk0) ? s0[r] : NEGV;
            float c = (mrow[r] && mk1) ? s1[r] : NEGV;
            float tmax = fmaxf(a, c);
#pragma unroll
            for (int off = 1; off < 16; off <<= 1)
                tmax = fmaxf(tmax, __shfl_xor(tmax, off, 32));
            float mn = fmaxf(mi[r], tmax);
            float sc = __expf(mi[r] - mn);
            float p0 = __expf(a - mn), p1 = __expf(c - mn);
            float rs = p0 + p1;
#pragma unroll
            for (int off = 1; off < 16; off <<= 1)
                rs += __shfl_xor(rs, off, 32);
            li[r] = li[r] * sc + rs;
            mi[r] = mn;
            o[0][r] *= sc; o[1][r] *= sc; o[2][r] *= sc; o[3][r] *= sc;
            lds[(r + 8 * half) * 32 + l16] = (__bf16)p0;
            lds[(r + 8 * half) * 32 + 16 + l16] = (__bf16)p1;
        }
        __syncthreads();
        union { v16bf v; v8bf hh[2]; } up;
        up.hh[0] = *(const v8bf*)(lds + l16 * 32 + half * 8);
        up.hh[1] = *(const v8bf*)(lds + l16 * 32 + 16 + half * 8);
        __syncthreads();
#pragma unroll
        for (int t4 = 0; t4 < 4; ++t4) {
            const __bf16* vrow = vbase + (size_t)(t4 * 16 + l16) * 1280 + kpos;
            o[t4] = wmma_bf16(up.v, load_bfrag(vrow, 0, half), o[t4]);
        }
    };

    const __bf16* kvb = Kv + (size_t)(b * 16 + h) * 1024 * 64;
    const __bf16* ktb = Kt + (size_t)(b * 16 + h) * 256 * 64;
    for (int it = 0; it < 32; ++it)
        tile(qf1, kvb, mkv + b * 1024, it * 32, it * 32);
    for (int it = 0; it < 8; ++it)
        tile(qf2, ktb, mkt + b * 256, it * 32, 1024 + it * 32);

#pragma unroll
    for (int r = 0; r < 8; ++r) {
        int qrow = q0 + r + 8 * half;
        float wn = 0.125f / li[r];   // 1/sqrt(64) applied post-softmax
#pragma unroll
        for (int t4 = 0; t4 < 4; ++t4)
            out[(size_t)(b * Lq + qrow) * 1024 + h * 64 + t4 * 16 + l16] =
                o[t4][r] * wn;
    }
}

extern "C" void kernel_launch(void* const* d_in, const int* in_sizes, int n_in,
                              void* d_out, int out_size, void* d_ws, size_t ws_size,
                              hipStream_t stream)
{
    const float* vid = (const float*)d_in[0];
    const unsigned char* vmask = (const unsigned char*)d_in[1];
    const float* txt = (const float*)d_in[2];
    const unsigned char* tmask = (const unsigned char*)d_in[3];
    // dict order: W_v2v, b_v2v, W_t2v, b_t2v, W_v2t, b_v2t, W_t2t, b_t2t
    const float* Wset[4] = {(const float*)d_in[4], (const float*)d_in[6],
                            (const float*)d_in[8], (const float*)d_in[10]};
    const float* Bset[4] = {(const float*)d_in[5], (const float*)d_in[7],
                            (const float*)d_in[9], (const float*)d_in[11]};

    const size_t D = 1024, Bn = 4, Lv = 1024, Lt = 256;
    const size_t NV = Bn * Lv * D;      // 4 Mi
    const size_t NT = Bn * Lt * D;      // 1 Mi
    const size_t NW = 3 * D * D;        // 3 Mi per weight set
    const size_t NVAL = Bn * 16 * 64 * (Lv + Lt); // 5 Mi per transposed V

    __bf16* p = (__bf16*)d_ws;
    __bf16* vidb = p; p += NV;
    __bf16* txtb = p; p += NT;
    __bf16* wb[4];
    for (int s = 0; s < 4; ++s) { wb[s] = p; p += NW; }
    __bf16* Qv1 = p; p += NV;
    __bf16* Qv2 = p; p += NV;
    __bf16* Kv1 = p; p += NV;
    __bf16* Kv2 = p; p += NV;
    __bf16* Qt1 = p; p += NT;
    __bf16* Qt2 = p; p += NT;
    __bf16* Kt1 = p; p += NT;
    __bf16* Kt2 = p; p += NT;
    __bf16* Vv  = p; p += NVAL;
    __bf16* Vtt = p; p += NVAL;

    cvt_kernel<<<dim3(NV / 256), 256, 0, stream>>>(vid, vidb, (int)NV);
    cvt_kernel<<<dim3(NT / 256), 256, 0, stream>>>(txt, txtb, (int)NT);
    for (int s = 0; s < 4; ++s)
        cvt_kernel<<<dim3(NW / 256), 256, 0, stream>>>(Wset[s], wb[s], (int)NW);

    auto W = [&](int s, int j) { return wb[s] + (size_t)j * D * D; };
    auto Bb = [&](int s, int j) { return Bset[s] + (size_t)j * D; };
    dim3 gv(4096 / 256, 16), gt(1024 / 256, 16);
    // sets: 0=v2v 1=t2v 2=v2t 3=t2t ; j: 0=q 1=k 2=v
    proj_gemm<<<gv, 256, 0, stream>>>(vidb, W(0, 0), Bb(0, 0), Qv1, 1024, 0, 0);
    proj_gemm<<<gv, 256, 0, stream>>>(vidb, W(0, 1), Bb(0, 1), Kv1, 1024, 0, 0);
    proj_gemm<<<gv, 256, 0, stream>>>(vidb, W(0, 2), Bb(0, 2), Vv,  1024, 1, 0);
    proj_gemm<<<gv, 256, 0, stream>>>(vidb, W(1, 0), Bb(1, 0), Qv2, 1024, 0, 0);
    proj_gemm<<<gv, 256, 0, stream>>>(vidb, W(2, 1), Bb(2, 1), Kv2, 1024, 0, 0);
    proj_gemm<<<gv, 256, 0, stream>>>(vidb, W(2, 2), Bb(2, 2), Vtt, 1024, 1, 0);
    proj_gemm<<<gt, 256, 0, stream>>>(txtb, W(1, 1), Bb(1, 1), Kt1, 256, 0, 0);
    proj_gemm<<<gt, 256, 0, stream>>>(txtb, W(1, 2), Bb(1, 2), Vv,  256, 1, 1024);
    proj_gemm<<<gt, 256, 0, stream>>>(txtb, W(2, 0), Bb(2, 0), Qt1, 256, 0, 0);
    proj_gemm<<<gt, 256, 0, stream>>>(txtb, W(3, 0), Bb(3, 0), Qt2, 256, 0, 0);
    proj_gemm<<<gt, 256, 0, stream>>>(txtb, W(3, 1), Bb(3, 1), Kt2, 256, 0, 0);
    proj_gemm<<<gt, 256, 0, stream>>>(txtb, W(3, 2), Bb(3, 2), Vtt, 256, 1, 1024);

    float* out = (float*)d_out;
    attn_kernel<<<dim3(1024 / 16, 16, 4), 32, 0, stream>>>(
        Qv1, Qv2, Kv1, Kt1, Vv, vmask, vmask, tmask, out, 1024);
    attn_kernel<<<dim3(256 / 16, 16, 4), 32, 0, stream>>>(
        Qt1, Qt2, Kv2, Kt2, Vtt, tmask, vmask, tmask, out + NV, 256);
}
